// VelDecoder_23733989278182
// MI455X (gfx1250) — matrix operand order repaired and checked by reading, not compiled, not run on hardware
//
#include <hip/hip_runtime.h>
#include <hip/hip_bf16.h>

// ---------------------------------------------------------------------------
// CDNA5 (gfx1250) persistent GRU + hierarchical limb decoder.
// wave32, WMMA f32_16x16x32_bf16, activations resident in LDS in A-fragment
// layout, weights repacked to bf16 B-fragment layout in workspace.
// ---------------------------------------------------------------------------

typedef __attribute__((ext_vector_type(16))) __bf16 v16bf;
typedef __attribute__((ext_vector_type(8)))  float  v8f;

#define H1 128
#define H2 256
#define H3 512
#define H4 1024
#define DPOSE 64
#define MTILE 32          // batch rows per workgroup (2 row-blocks of 16)
#define NTHREADS 256      // 8 waves

// ---- LDS layout (units: ushort halves) ----
// A-fragment buffers: [rb(2)][kt(KT)][lane(32)][e(16)]
#define HS0_OFF   0            // 2*32*512 = 32768 halves (hs double buffer A)
#define HS1_OFF   32768        // 32768
#define ACTB_OFF  65536        // 32768  (fb, then w2 outputs)
#define ACTC_OFF  98304        // 32768  (ub | lb)
#define XA_OFF    131072       // 2*2*512 = 2048 (x in A-layout, K=64)
#define F32_OFF   133120       // float region starts here (half-offset)
// floats: xf[32*64]=2048, limbF[8*32*16]=4096  -> 6144 floats = 12288 halves
#define SMEM_HALVES (133120 + 12288)
#define SMEM_BYTES  (SMEM_HALVES * 2)   // 290816 B < 320 KB/WGP

// ---------------- bf16 helpers ----------------
static __device__ __forceinline__ unsigned short f2bf(float f) {
  union { float f; unsigned int u; } v; v.f = f;
  unsigned int u = v.u;
  unsigned int r = u + 0x7FFFu + ((u >> 16) & 1u);   // RNE
  return (unsigned short)(r >> 16);
}
static __device__ __forceinline__ float bf2f(unsigned short h) {
  union { unsigned int u; float f; } v; v.u = ((unsigned int)h) << 16;
  return v.f;
}
static __device__ __forceinline__ float sigf(float x) {
  return 1.0f / (1.0f + __expf(-x));
}

static __device__ __forceinline__ v8f wmma_bf16(v16bf a, v16bf b, v8f c) {
  return __builtin_amdgcn_wmma_f32_16x16x32_bf16(
      false, a, false, b, (short)0, c, false, false);
}

// ---- fragment accessors ----
// B fragment (16x32 bf16 weight tile, fragment-major in global memory)
static __device__ __forceinline__ v16bf loadB(const unsigned short* w, int frag,
                                              int lane) {
  return *reinterpret_cast<const v16bf*>(w + (((size_t)frag) << 9) + (lane << 4));
}
// A fragment from LDS A-layout buffer
static __device__ __forceinline__ v16bf loadA(const unsigned short* sm, int base,
                                              int KT, int rb, int kt, int lane) {
  return *reinterpret_cast<const v16bf*>(
      sm + base + ((((rb * KT + kt) << 5) + lane) << 4));
}
// scalar element address in A-layout: row in [0,32), col = K index
static __device__ __forceinline__ int aAddr(int base, int KT, int row, int col) {
  int rb = row >> 4;
  int kt = col >> 5;
  int kl = col & 31;
  int lane_t = (row & 15) + (((kl & 15) >= 8) ? 16 : 0);
  int e = (kl & 7) + ((kl >= 16) ? 8 : 0);
  return base + (((rb * KT + kt) << 5) + lane_t) * 16 + e;
}
static __device__ __forceinline__ void storeA(unsigned short* sm, int base, int KT,
                                              int row, int col, float v) {
  sm[aAddr(base, KT, row, col)] = f2bf(v);
}
static __device__ __forceinline__ float readA(const unsigned short* sm, int base,
                                              int KT, int row, int col) {
  return bf2f(sm[aAddr(base, KT, row, col)]);
}

// ---------------- weight repack kernel ----------------
// src: float [N, K] row-major (out-features x in-features, i.e. x @ W^T).
// dst: bf16 fragment-major: frag(nt,kt) = 512 halves; lane=col%16 (+16 for K hi),
//      element e -> K = kt*32 + (e>=8?16:0) + (lane>=16?8:0) + (e&7).  N padded to mult of 16.
__global__ void pack_weight_bf16(const float* __restrict__ src,
                                 unsigned short* __restrict__ dst,
                                 int N, int K) {
  int KT = K >> 5;
  int NT = (N + 15) >> 4;
  int total = NT * KT * 512;
  for (int idx = blockIdx.x * blockDim.x + threadIdx.x; idx < total;
       idx += gridDim.x * blockDim.x) {
    int e = idx & 15;
    int lane = (idx >> 4) & 31;
    int frag = idx >> 9;
    int kt = frag % KT;
    int nt = frag / KT;
    int col = (nt << 4) + (lane & 15);
    int k = (kt << 5) + ((e & 8) << 1) + ((lane >> 4) << 3) + (e & 7);
    float v = (col < N) ? src[(size_t)col * K + k] : 0.0f;
    dst[idx] = f2bf(v);
  }
}

// ---------------- main persistent kernel ----------------
struct VDParams {
  const float* h;
  const float* gt;
  const float *b_ih, *b_hh, *b4, *b3_arm, *b3_leg;
  const float *b2_rarm, *b2_larm, *b2_rleg, *b2_lleg;
  const float *b1_rarm, *b1_larm, *b1_rleg, *b1_lleg, *b1_torso;
  const int* steps;
  float* out;
  const unsigned short *Wih, *Whh, *W4, *W3a, *W3l;
  const unsigned short *W2ra, *W2la, *W2rl, *W2ll;
  const unsigned short *W1ra, *W1la, *W1rl, *W1ll, *W1t;
};

__global__ __launch_bounds__(NTHREADS)
void veldecoder_persistent(VDParams P) {
  extern __shared__ __align__(32) unsigned short sm[];
  float* fbase = reinterpret_cast<float*>(sm + F32_OFF);
  float* xf = fbase;                // [32][64]
  float* limbF = fbase + 2048;      // [8][32][16]

  const int lane = threadIdx.x & 31;
  const int wave = threadIdx.x >> 5;
  const int bbase = blockIdx.x * MTILE;
  const int nsteps = *P.steps;      // = T = 128

  // ---- init: hs[0..31][0..1023] -> HS0 (bf16 A-layout), x = gt[:,0,:] ----
  for (int idx = threadIdx.x; idx < MTILE * H4; idx += NTHREADS) {
    int row = idx >> 10, col = idx & 1023;
    storeA(sm, HS0_OFF, 32, row, col, P.h[(size_t)(bbase + row) * H4 + col]);
  }
  for (int idx = threadIdx.x; idx < MTILE * DPOSE; idx += NTHREADS) {
    int row = idx >> 6, d = idx & 63;
    float v = P.gt[((size_t)(bbase + row) * nsteps + 0) * DPOSE + d];
    xf[idx] = v;
    storeA(sm, XA_OFF, 2, row, d, v);
  }
  __syncthreads();

  int cur = 0;
  for (int s = 0; s < nsteps; ++s) {
    const int hsR = cur ? HS1_OFF : HS0_OFF;
    const int hsW = cur ? HS0_OFF : HS1_OFF;

    // ===== Phase 1: GRU gates; hs ntiles 0..63, 8 per wave =====
    for (int i = 0; i < 8; ++i) {
      const int t = wave + 8 * i;
      v8f ar0 = {}, ar1 = {}, az0 = {}, az1 = {};
      v8f an0 = {}, an1 = {}, ah0 = {}, ah1 = {};
      // x @ W_ih^T  (K = 64)
#pragma unroll
      for (int kt = 0; kt < 2; ++kt) {
        v16bf a0 = loadA(sm, XA_OFF, 2, 0, kt, lane);
        v16bf a1 = loadA(sm, XA_OFF, 2, 1, kt, lane);
        v16bf br = loadB(P.Wih, (t)*2 + kt, lane);
        v16bf bz = loadB(P.Wih, (64 + t) * 2 + kt, lane);
        v16bf bn = loadB(P.Wih, (128 + t) * 2 + kt, lane);
        ar0 = wmma_bf16(a0, br, ar0); ar1 = wmma_bf16(a1, br, ar1);
        az0 = wmma_bf16(a0, bz, az0); az1 = wmma_bf16(a1, bz, az1);
        an0 = wmma_bf16(a0, bn, an0); an1 = wmma_bf16(a1, bn, an1);
      }
      // hs @ W_hh^T  (K = 1024)
      for (int kt = 0; kt < 32; ++kt) {
        if (kt + 1 < 32)
          __builtin_prefetch(P.Whh + (((size_t)(t * 32 + kt + 1)) << 9), 0, 1);
        v16bf a0 = loadA(sm, hsR, 32, 0, kt, lane);
        v16bf a1 = loadA(sm, hsR, 32, 1, kt, lane);
        v16bf br = loadB(P.Whh, t * 32 + kt, lane);
        v16bf bz = loadB(P.Whh, (64 + t) * 32 + kt, lane);
        v16bf bh = loadB(P.Whh, (128 + t) * 32 + kt, lane);
        ar0 = wmma_bf16(a0, br, ar0); ar1 = wmma_bf16(a1, br, ar1);
        az0 = wmma_bf16(a0, bz, az0); az1 = wmma_bf16(a1, bz, az1);
        ah0 = wmma_bf16(a0, bh, ah0); ah1 = wmma_bf16(a1, bh, ah1);
      }
      // gate nonlinearity + hs update (C layout: VGPR e -> row e / e+8)
#pragma unroll
      for (int rb = 0; rb < 2; ++rb) {
        v8f ar = rb ? ar1 : ar0, az = rb ? az1 : az0;
        v8f an = rb ? an1 : an0, ah = rb ? ah1 : ah0;
#pragma unroll
        for (int e = 0; e < 8; ++e) {
          int row = rb * 16 + e + ((lane >= 16) ? 8 : 0);
          int col = t * 16 + (lane & 15);
          float rr = sigf(ar[e] + P.b_ih[col] + P.b_hh[col]);
          float zz = sigf(az[e] + P.b_ih[H4 + col] + P.b_hh[H4 + col]);
          float hn = ah[e] + P.b_hh[2 * H4 + col];
          float nn = tanhf(an[e] + P.b_ih[2 * H4 + col] + rr * hn);
          float hold = readA(sm, hsR, 32, row, col);
          storeA(sm, hsW, 32, row, col, (1.0f - zz) * nn + zz * hold);
        }
      }
    }
    __syncthreads();
    cur ^= 1;
    const int hsC = cur ? HS1_OFF : HS0_OFF;

    // ===== Phase 2: fb = hs @ w4^T + b4  -> ACTB (64 ntiles) =====
    for (int i = 0; i < 8; ++i) {
      const int t = wave + 8 * i;
      v8f c0 = {}, c1 = {};
      for (int kt = 0; kt < 32; ++kt) {
        v16bf a0 = loadA(sm, hsC, 32, 0, kt, lane);
        v16bf a1 = loadA(sm, hsC, 32, 1, kt, lane);
        v16bf b = loadB(P.W4, t * 32 + kt, lane);
        c0 = wmma_bf16(a0, b, c0); c1 = wmma_bf16(a1, b, c1);
      }
#pragma unroll
      for (int rb = 0; rb < 2; ++rb) {
        v8f c = rb ? c1 : c0;
#pragma unroll
        for (int e = 0; e < 8; ++e) {
          int row = rb * 16 + e + ((lane >= 16) ? 8 : 0);
          int col = t * 16 + (lane & 15);
          storeA(sm, ACTB_OFF, 32, row, col, c[e] + P.b4[col]);
        }
      }
    }
    __syncthreads();

    // ===== Phase 3: ub = fb[:, :512] @ w3_arm^T; lb = fb[:,512:] @ w3_leg^T =====
    for (int i = 0; i < 8; ++i) {
      const int o = wave + 8 * i;          // 0..63 ; <32 arm, >=32 leg
      const bool arm = (o < 32);
      const int nl = arm ? o : (o - 32);
      const unsigned short* W = arm ? P.W3a : P.W3l;
      const float* bias = arm ? P.b3_arm : P.b3_leg;
      const int ktoff = arm ? 0 : 16;
      v8f c0 = {}, c1 = {};
      for (int kt = 0; kt < 16; ++kt) {
        v16bf a0 = loadA(sm, ACTB_OFF, 32, 0, ktoff + kt, lane);
        v16bf a1 = loadA(sm, ACTB_OFF, 32, 1, ktoff + kt, lane);
        v16bf b = loadB(W, nl * 16 + kt, lane);
        c0 = wmma_bf16(a0, b, c0); c1 = wmma_bf16(a1, b, c1);
      }
#pragma unroll
      for (int rb = 0; rb < 2; ++rb) {
        v8f c = rb ? c1 : c0;
#pragma unroll
        for (int e = 0; e < 8; ++e) {
          int row = rb * 16 + e + ((lane >= 16) ? 8 : 0);
          int cl = nl * 16 + (lane & 15);
          storeA(sm, ACTC_OFF, 32, row, o * 16 + (lane & 15), c[e] + bias[cl]);
        }
      }
    }
    __syncthreads();

    // ===== Phase 4: ra2/la2/rl2/ll2 (4x [32,256]) -> ACTB =====
    for (int i = 0; i < 8; ++i) {
      const int o = wave + 8 * i;          // 0..63
      const int limb = o >> 4;             // 0 ra2, 1 la2, 2 rl2, 3 ll2
      const int nl = o & 15;
      const unsigned short* W; const float* bias;
      switch (limb) {
        case 0:  W = P.W2ra; bias = P.b2_rarm; break;
        case 1:  W = P.W2la; bias = P.b2_larm; break;
        case 2:  W = P.W2rl; bias = P.b2_rleg; break;
        default: W = P.W2ll; bias = P.b2_lleg; break;
      }
      const int ktoff = limb * 8;
      v8f c0 = {}, c1 = {};
      for (int kt = 0; kt < 8; ++kt) {
        v16bf a0 = loadA(sm, ACTC_OFF, 32, 0, ktoff + kt, lane);
        v16bf a1 = loadA(sm, ACTC_OFF, 32, 1, ktoff + kt, lane);
        v16bf b = loadB(W, nl * 8 + kt, lane);
        c0 = wmma_bf16(a0, b, c0); c1 = wmma_bf16(a1, b, c1);
      }
#pragma unroll
      for (int rb = 0; rb < 2; ++rb) {
        v8f c = rb ? c1 : c0;
#pragma unroll
        for (int e = 0; e < 8; ++e) {
          int row = rb * 16 + e + ((lane >= 16) ? 8 : 0);
          int cl = nl * 16 + (lane & 15);
          storeA(sm, ACTB_OFF, 32, row, o * 16 + (lane & 15), c[e] + bias[cl]);
        }
      }
    }
    __syncthreads();

    // ===== Phase 5: eight tiny heads, one per wave (N padded to 16, K=128) ====
    {
      const unsigned short* W; const float* bias; int nvalid;
      switch (wave) {
        case 0:  W = P.W1ra; bias = P.b1_rarm;  nvalid = 9;  break; // ra1
        case 1:  W = P.W1t;  bias = P.b1_torso; nvalid = 16; break; // t1
        case 2:  W = P.W1la; bias = P.b1_larm;  nvalid = 9;  break; // la1
        case 3:  W = P.W1t;  bias = P.b1_torso; nvalid = 16; break; // t2
        case 4:  W = P.W1rl; bias = P.b1_rleg;  nvalid = 15; break; // rl1
        case 5:  W = P.W1t;  bias = P.b1_torso; nvalid = 16; break; // t3
        case 6:  W = P.W1ll; bias = P.b1_lleg;  nvalid = 15; break; // ll1
        default: W = P.W1t;  bias = P.b1_torso; nvalid = 16; break; // t4
      }
      const int ktoff = wave * 4;
      v8f c0 = {}, c1 = {};
#pragma unroll
      for (int kt = 0; kt < 4; ++kt) {
        v16bf a0 = loadA(sm, ACTB_OFF, 32, 0, ktoff + kt, lane);
        v16bf a1 = loadA(sm, ACTB_OFF, 32, 1, ktoff + kt, lane);
        v16bf b = loadB(W, kt, lane);
        c0 = wmma_bf16(a0, b, c0); c1 = wmma_bf16(a1, b, c1);
      }
#pragma unroll
      for (int rb = 0; rb < 2; ++rb) {
        v8f c = rb ? c1 : c0;
#pragma unroll
        for (int e = 0; e < 8; ++e) {
          int row = rb * 16 + e + ((lane >= 16) ? 8 : 0);
          int cl = lane & 15;
          float v = (cl < nvalid) ? (c[e] + bias[cl]) : 0.0f;
          limbF[wave * 512 + row * 16 + cl] = v;
        }
      }
    }
    __syncthreads();

    // ===== Phase 6: assemble pose, residual, emit, refresh x fragments =====
    for (int idx = threadIdx.x; idx < MTILE * DPOSE; idx += NTHREADS) {
      int row = idx >> 6, d = idx & 63;
      int rb16 = row * 16;
      float val;
      if (d < 13) {
        val = 0.25f * (limbF[1 * 512 + rb16 + d] + limbF[3 * 512 + rb16 + d] +
                       limbF[5 * 512 + rb16 + d] + limbF[7 * 512 + rb16 + d]);
      } else if (d < 22) {        // la1 (9)
        val = limbF[2 * 512 + rb16 + (d - 13)];
      } else if (d < 31) {        // ra1 (9)
        val = limbF[0 * 512 + rb16 + (d - 22)];
      } else if (d < 46) {        // ll1 (15)
        val = limbF[6 * 512 + rb16 + (d - 31)];
      } else if (d < 61) {        // rl1 (15)
        val = limbF[4 * 512 + rb16 + (d - 46)];
      } else {                    // torso[13:16]
        int c = 13 + (d - 61);
        val = 0.25f * (limbF[1 * 512 + rb16 + c] + limbF[3 * 512 + rb16 + c] +
                       limbF[5 * 512 + rb16 + c] + limbF[7 * 512 + rb16 + c]);
      }
      float xn = val + xf[idx];
      xf[idx] = xn;
      P.out[((size_t)(bbase + row) * nsteps + s) * DPOSE + d] = xn;
      storeA(sm, XA_OFF, 2, row, d, xn);
    }
    __syncthreads();
  }
}

// ---------------------------------------------------------------------------
extern "C" void kernel_launch(void* const* d_in, const int* in_sizes, int n_in,
                              void* d_out, int out_size, void* d_ws, size_t ws_size,
                              hipStream_t stream) {
  (void)in_sizes; (void)n_in; (void)out_size; (void)ws_size;
  const float* h       = (const float*)d_in[0];
  const float* gt      = (const float*)d_in[1];
  const float* W_ih    = (const float*)d_in[2];
  const float* W_hh    = (const float*)d_in[3];
  const float* b_ih    = (const float*)d_in[4];
  const float* b_hh    = (const float*)d_in[5];
  const float* w4      = (const float*)d_in[6];
  const float* b4      = (const float*)d_in[7];
  const float* w3_arm  = (const float*)d_in[8];
  const float* b3_arm  = (const float*)d_in[9];
  const float* w3_leg  = (const float*)d_in[10];
  const float* b3_leg  = (const float*)d_in[11];
  const float* w2_rarm = (const float*)d_in[12];
  const float* b2_rarm = (const float*)d_in[13];
  const float* w2_larm = (const float*)d_in[14];
  const float* b2_larm = (const float*)d_in[15];
  const float* w2_rleg = (const float*)d_in[16];
  const float* b2_rleg = (const float*)d_in[17];
  const float* w2_lleg = (const float*)d_in[18];
  const float* b2_lleg = (const float*)d_in[19];
  const float* w1_rarm = (const float*)d_in[20];
  const float* b1_rarm = (const float*)d_in[21];
  const float* w1_larm = (const float*)d_in[22];
  const float* b1_larm = (const float*)d_in[23];
  const float* w1_rleg = (const float*)d_in[24];
  const float* b1_rleg = (const float*)d_in[25];
  const float* w1_lleg = (const float*)d_in[26];
  const float* b1_lleg = (const float*)d_in[27];
  const float* w1_torso= (const float*)d_in[28];
  const float* b1_torso= (const float*)d_in[29];
  const int*   steps   = (const int*)d_in[30];

  unsigned short* ws = (unsigned short*)d_ws;
  size_t off = 0;
  auto alloc = [&](int N, int K) {
    unsigned short* p = ws + off;
    off += (size_t)((N + 15) / 16) * (K / 32) * 512;
    return p;
  };
  unsigned short* pWih = alloc(3 * H4, DPOSE);
  unsigned short* pWhh = alloc(3 * H4, H4);
  unsigned short* pW4  = alloc(2 * H3, H4);
  unsigned short* pW3a = alloc(2 * H2, H3);
  unsigned short* pW3l = alloc(2 * H2, H3);
  unsigned short* pW2ra= alloc(2 * H1, H2);
  unsigned short* pW2la= alloc(2 * H1, H2);
  unsigned short* pW2rl= alloc(2 * H1, H2);
  unsigned short* pW2ll= alloc(2 * H1, H2);
  unsigned short* pW1ra= alloc(9,  H1);
  unsigned short* pW1la= alloc(9,  H1);
  unsigned short* pW1rl= alloc(15, H1);
  unsigned short* pW1ll= alloc(15, H1);
  unsigned short* pW1t = alloc(16, H1);

  auto pack = [&](const float* src, unsigned short* dst, int N, int K) {
    int total = ((N + 15) / 16) * (K / 32) * 512;
    int blocks = (total + 255) / 256;
    if (blocks > 1024) blocks = 1024;
    hipLaunchKernelGGL(pack_weight_bf16, dim3(blocks), dim3(256), 0, stream,
                       src, dst, N, K);
  };
  pack(W_ih,   pWih,  3 * H4, DPOSE);
  pack(W_hh,   pWhh,  3 * H4, H4);
  pack(w4,     pW4,   2 * H3, H4);
  pack(w3_arm, pW3a,  2 * H2, H3);
  pack(w3_leg, pW3l,  2 * H2, H3);
  pack(w2_rarm,pW2ra, 2 * H1, H2);
  pack(w2_larm,pW2la, 2 * H1, H2);
  pack(w2_rleg,pW2rl, 2 * H1, H2);
  pack(w2_lleg,pW2ll, 2 * H1, H2);
  pack(w1_rarm,pW1ra, 9,  H1);
  pack(w1_larm,pW1la, 9,  H1);
  pack(w1_rleg,pW1rl, 15, H1);
  pack(w1_lleg,pW1ll, 15, H1);
  pack(w1_torso,pW1t, 16, H1);

  VDParams P;
  P.h = h; P.gt = gt;
  P.b_ih = b_ih; P.b_hh = b_hh; P.b4 = b4;
  P.b3_arm = b3_arm; P.b3_leg = b3_leg;
  P.b2_rarm = b2_rarm; P.b2_larm = b2_larm;
  P.b2_rleg = b2_rleg; P.b2_lleg = b2_lleg;
  P.b1_rarm = b1_rarm; P.b1_larm = b1_larm;
  P.b1_rleg = b1_rleg; P.b1_lleg = b1_lleg; P.b1_torso = b1_torso;
  P.steps = steps; P.out = (float*)d_out;
  P.Wih = pWih; P.Whh = pWhh; P.W4 = pW4; P.W3a = pW3a; P.W3l = pW3l;
  P.W2ra = pW2ra; P.W2la = pW2la; P.W2rl = pW2rl; P.W2ll = pW2ll;
  P.W1ra = pW1ra; P.W1la = pW1la; P.W1rl = pW1rl; P.W1ll = pW1ll; P.W1t = pW1t;

  const int nWG = 4096 / MTILE;   // 128 workgroups, 8 waves each
  hipLaunchKernelGGL(veldecoder_persistent, dim3(nWG), dim3(NTHREADS),
                     SMEM_BYTES, stream, P);
}
